// MultiHeadedAttentionTree_52647709114821
// MI455X (gfx1250) — matrix operand order repaired and checked by reading, not compiled, not run on hardware
//
#include <hip/hip_runtime.h>
#include <hip/hip_bf16.h>

// ---------------------------------------------------------------------------
// Relative-position multi-head attention, bf16 WMMA pipeline for gfx1250.
// B=4, S=512, D=512, H=8, dk=64, NUM_REL=41.
// GEMMs: TDM-staged (tensor_load_to_lds) double-buffered 128x128 block tiles.
// ---------------------------------------------------------------------------

typedef __attribute__((ext_vector_type(16))) __bf16 v16bf;
typedef __attribute__((ext_vector_type(8)))  __bf16 v8bf;
typedef __attribute__((ext_vector_type(8)))  float  v8f;
typedef __attribute__((ext_vector_type(4)))  unsigned int u32x4;
typedef __attribute__((ext_vector_type(8)))  int i32x8;
typedef __attribute__((ext_vector_type(4)))  int i32x4;

#define BATCH  4
#define SEQ    512
#define DMODEL 512
#define NHEAD  8
#define DKH    64
#define NREL   41
#define NRELP  48
#define NEGV   (-1.0e9f)

#if defined(__has_builtin)
#if __has_builtin(__builtin_amdgcn_tensor_load_to_lds) && \
    __has_builtin(__builtin_amdgcn_s_wait_tensorcnt)
#define HAVE_TDM 1
#endif
#endif
#ifndef HAVE_TDM
#define HAVE_TDM 0
#endif

// ---------------------------------------------------------------------------
// WMMA fragment loaders (CDNA5 16-bit VGPR layouts, wave32).
// A (16xK row-major, ld): lane l -> M = l&15, K chunks {kf..kf+7, kf+16..kf+23},
//   kf = (l&16)?8:0.
// B (Kx16 stored N-major: (k,n) at base[n*ld+k]): lane l -> N = l&15,
//   K chunks {kf..kf+7, kf+16..kf+23}, kf = (l&16)?8:0.
// C/D f32: lane l -> N = l&15, vgpr r -> M = 8*(l>>4)+r.
// ---------------------------------------------------------------------------
__device__ __forceinline__ v16bf frag_a(const __bf16* base, int ld) {
  const int l  = threadIdx.x & 31;
  const int m  = l & 15;
  const int kf = (l & 16) ? 8 : 0;
  const __bf16* p = base + m * ld + kf;
  v8bf c0 = *(const v8bf*)(p);
  v8bf c1 = *(const v8bf*)(p + 16);
  v16bf r;
#pragma unroll
  for (int i = 0; i < 8; ++i) { r[i] = c0[i]; r[i + 8] = c1[i]; }
  return r;
}

__device__ __forceinline__ v16bf frag_b(const __bf16* base, int ld) {
  const int l  = threadIdx.x & 31;
  const int n  = l & 15;
  const int kc = (l & 16) ? 8 : 0;
  const __bf16* p = base + n * ld + kc;
  v8bf c0 = *(const v8bf*)(p);
  v8bf c1 = *(const v8bf*)(p + 16);
  v16bf r;
#pragma unroll
  for (int i = 0; i < 8; ++i) { r[i] = c0[i]; r[i + 8] = c1[i]; }
  return r;
}

__device__ __forceinline__ v8f wmma_bf16(v16bf a, v16bf b, v8f c) {
  return __builtin_amdgcn_wmma_f32_16x16x32_bf16(
      false, a, false, b, (short)0, c, false, false);
}

// ---------------------------------------------------------------------------
// TDM: issue a 2D tile load (bf16 elements) global -> LDS.
// D# packing per CDNA5 ISA:
//   group0: [1:0]=count=1, [63:32]=lds_addr, [120:64]=global_addr, [127:126]=2
//   group1: [17:16]=data_size(1 => 2B), [79:48]=tensor_dim0, [111:80]=tensor_dim1,
//           [127:112]=tile_dim0, [143:128]=tile_dim1, [207:160]=dim0_stride
// ---------------------------------------------------------------------------
#if HAVE_TDM
__device__ __forceinline__ void tdm_load_2d(unsigned lds_addr, const void* gptr,
                                            unsigned tensor_d0, unsigned tensor_d1,
                                            unsigned tile_d0, unsigned tile_d1,
                                            unsigned long long stride0_elems) {
  unsigned long long ga = (unsigned long long)gptr;
  u32x4 g0;
  g0[0] = 1u;                                             // count=1, user desc
  g0[1] = lds_addr;                                       // LDS byte address
  g0[2] = (unsigned)(ga & 0xffffffffu);                   // global_addr[31:0]
  g0[3] = (unsigned)((ga >> 32) & 0x1ffffffu) | (2u << 30); // addr[56:32] | type=2
  i32x8 g1;
  g1[0] = (int)(1u << 16);                                // data_size=1 (2 bytes)
  g1[1] = (int)((tensor_d0 & 0xffffu) << 16);             // tensor_dim0[15:0]
  g1[2] = (int)(((tensor_d0 >> 16) & 0xffffu) |
                ((tensor_d1 & 0xffffu) << 16));           // td0[31:16] | td1[15:0]
  g1[3] = (int)(((tensor_d1 >> 16) & 0xffffu) |
                ((tile_d0 & 0xffffu) << 16));             // td1[31:16] | tile_d0
  g1[4] = (int)(tile_d1 & 0xffffu);                       // tile_d1 (tile_d2 = 0)
  g1[5] = (int)(unsigned)(stride0_elems & 0xffffffffu);   // dim0_stride[31:0]
  g1[6] = (int)(unsigned)((stride0_elems >> 32) & 0xffffu); // dim0_stride[47:32]
  g1[7] = 0;
  i32x4 z4 = {0, 0, 0, 0};
#if __clang_major__ >= 23
  i32x8 z8 = {0, 0, 0, 0, 0, 0, 0, 0};
  __builtin_amdgcn_tensor_load_to_lds(g0, g1, z4, z4, z8, 0);
#else
  __builtin_amdgcn_tensor_load_to_lds(g0, g1, z4, z4, 0);
#endif
}
#endif

__device__ __forceinline__ unsigned lds_byte_addr(const void* p) {
  // generic pointers into the LDS aperture carry the LDS offset in addr[31:0]
  return (unsigned)(unsigned long long)p;
}

// ---------------------------------------------------------------------------
// Conversion / layout-prep kernels
// ---------------------------------------------------------------------------
__global__ void cvt_bf16_kernel(const float* __restrict__ src,
                                __bf16* __restrict__ dst, int n) {
  int i = blockIdx.x * blockDim.x + threadIdx.x;
  if (i < n) dst[i] = (__bf16)src[i];
}

// Wt[n][k] = W[k][n]  (W is [K=512][N=512] f32, Wt is [N][K] bf16)
__global__ void transpose_w_kernel(const float* __restrict__ W,
                                   __bf16* __restrict__ Wt) {
  int n = blockIdx.x;
  for (int k = threadIdx.x; k < DMODEL; k += blockDim.x)
    Wt[(size_t)n * DMODEL + k] = (__bf16)W[(size_t)k * DMODEL + n];
}

__global__ void emb_pad_kernel(const float* __restrict__ embk,
                               __bf16* __restrict__ out) {
  int i = blockIdx.x * blockDim.x + threadIdx.x;
  if (i < NRELP * DKH) {
    int r = i >> 6;
    out[i] = (r < NREL) ? (__bf16)embk[i] : (__bf16)0.0f;
  }
}

// ---------------------------------------------------------------------------
// Tiled GEMM: C[m,n] = sum_k A[m,k]*Bt[n,k] + bias[n]
// 256 threads = 8 waves; block tile 128x128; wave tile 64x32 (4x2 WMMA tiles).
// A and B panels (128x32 bf16) staged to LDS by the Tensor Data Mover,
// double-buffered, published via s_wait_tensorcnt + workgroup barrier.
// MODE 0: bf16 head-split [b,h,s,d] | MODE 1: bf16 [b,h,d,s] | MODE 2: f32 [m,n]
// ---------------------------------------------------------------------------
template <int MODE>
__global__ __launch_bounds__(256) void gemm_tile_kernel(
    const __bf16* __restrict__ A, const __bf16* __restrict__ Bt,
    const float* __restrict__ bias, void* __restrict__ out,
    int Mdim, int Ndim, int Kdim) {
  __shared__ __align__(16) __bf16 Al[2][128 * 32];   // 16 KB
  __shared__ __align__(16) __bf16 Bl[2][128 * 32];   // 16 KB

  const int tid  = threadIdx.x;
  const int wave = tid >> 5;
  const int l    = tid & 31;
  const int wm   = wave >> 2;           // 0..1 -> 64-row slab
  const int wn   = wave & 3;            // 0..3 -> 32-col slab
  const int m0   = blockIdx.y * 128;
  const int n0   = blockIdx.x * 128;
  const int KSTEPS = Kdim / 32;

  v8f acc[4][2] = {};

#if HAVE_TDM
  // ---- TDM double-buffered pipeline ----
  if (wave == 0) {
    tdm_load_2d(lds_byte_addr(&Al[0][0]), A + (size_t)m0 * Kdim,
                (unsigned)Kdim, 128u, 32u, 128u, (unsigned long long)Kdim);
    tdm_load_2d(lds_byte_addr(&Bl[0][0]), Bt + (size_t)n0 * Kdim,
                (unsigned)Kdim, 128u, 32u, 128u, (unsigned long long)Kdim);
    __builtin_amdgcn_s_wait_tensorcnt(0);
  }
  __syncthreads();
  for (int ks = 0; ks < KSTEPS; ++ks) {
    const int cur = ks & 1, nxt = cur ^ 1;
    if (wave == 0 && ks + 1 < KSTEPS) {
      tdm_load_2d(lds_byte_addr(&Al[nxt][0]),
                  A + (size_t)m0 * Kdim + (ks + 1) * 32,
                  (unsigned)Kdim, 128u, 32u, 128u, (unsigned long long)Kdim);
      tdm_load_2d(lds_byte_addr(&Bl[nxt][0]),
                  Bt + (size_t)n0 * Kdim + (ks + 1) * 32,
                  (unsigned)Kdim, 128u, 32u, 128u, (unsigned long long)Kdim);
    }
    v16bf af[4], bf[2];
#pragma unroll
    for (int i = 0; i < 4; ++i)
      af[i] = frag_a(&Al[cur][(wm * 64 + i * 16) * 32], 32);
#pragma unroll
    for (int j = 0; j < 2; ++j)
      bf[j] = frag_b(&Bl[cur][(wn * 32 + j * 16) * 32], 32);
#pragma unroll
    for (int i = 0; i < 4; ++i)
#pragma unroll
      for (int j = 0; j < 2; ++j)
        acc[i][j] = wmma_bf16(af[i], bf[j], acc[i][j]);
    if (wave == 0 && ks + 1 < KSTEPS) __builtin_amdgcn_s_wait_tensorcnt(0);
    __syncthreads();
  }
#else
  // ---- cooperative synchronous staging fallback ----
  for (int ks = 0; ks < KSTEPS; ++ks) {
    __syncthreads();
    for (int idx = tid; idx < 512; idx += 256) {  // 128 rows x 4 chunks of 16B
      const int row = idx >> 2, c = (idx & 3) * 8;
      *(v8bf*)(&Al[0][row * 32 + c]) =
          *(const v8bf*)(A + ((size_t)m0 + row) * Kdim + ks * 32 + c);
      *(v8bf*)(&Bl[0][row * 32 + c]) =
          *(const v8bf*)(Bt + ((size_t)n0 + row) * Kdim + ks * 32 + c);
    }
    __syncthreads();
    v16bf af[4], bf[2];
#pragma unroll
    for (int i = 0; i < 4; ++i)
      af[i] = frag_a(&Al[0][(wm * 64 + i * 16) * 32], 32);
#pragma unroll
    for (int j = 0; j < 2; ++j)
      bf[j] = frag_b(&Bl[0][(wn * 32 + j * 16) * 32], 32);
#pragma unroll
    for (int i = 0; i < 4; ++i)
#pragma unroll
      for (int j = 0; j < 2; ++j)
        acc[i][j] = wmma_bf16(af[i], bf[j], acc[i][j]);
  }
#endif

  // ---- epilogue ----
#pragma unroll
  for (int i = 0; i < 4; ++i) {
#pragma unroll
    for (int j = 0; j < 2; ++j) {
      const int mbase = m0 + wm * 64 + i * 16;
      const int nbase = n0 + wn * 32 + j * 16;
      const int n = nbase + (l & 15);
      const float bb = bias[n];
#pragma unroll
      for (int r = 0; r < 8; ++r) {
        const int m = mbase + r + ((l >> 4) << 3);
        const float v = acc[i][j][r] + bb;
        if (MODE == 2) {
          ((float*)out)[(size_t)m * Ndim + n] = v;
        } else {
          const int b_ = m >> 9, s_ = m & 511;
          const int h_ = n >> 6, d_ = n & 63;
          const __bf16 bv = (__bf16)v;
          if (MODE == 0)
            ((__bf16*)out)[(((size_t)(b_ * NHEAD + h_) * SEQ + s_) * DKH) + d_] = bv;
          else
            ((__bf16*)out)[(((size_t)(b_ * NHEAD + h_) * DKH + d_) * SEQ) + s_] = bv;
        }
      }
    }
  }
}

// ---------------------------------------------------------------------------
// qe[b,h,s,r] = sum_d q[b,h,s,d] * emb_k[r,d]   (r padded to 48)
// ---------------------------------------------------------------------------
__global__ __launch_bounds__(32) void qe_kernel(
    const __bf16* __restrict__ qb, const __bf16* __restrict__ embkb,
    float* __restrict__ qe) {
  const int nt = blockIdx.x;   // 0..2
  const int mt = blockIdx.y;   // 0..(B*H*S/16 - 1)
  const __bf16* arow = qb    + (size_t)mt * 16 * DKH;
  const __bf16* brow = embkb + (size_t)nt * 16 * DKH;
  v8f acc = {};
#pragma unroll
  for (int k = 0; k < DKH; k += 32) {
    v16bf a = frag_a(arow + k, DKH);
    v16bf b = frag_b(brow + k, DKH);
    acc = wmma_bf16(a, b, acc);
  }
  const int l = threadIdx.x & 31;
  const int n = nt * 16 + (l & 15);
#pragma unroll
  for (int r = 0; r < 8; ++r) {
    const int m = mt * 16 + r + ((l >> 4) << 3);
    qe[(size_t)m * NRELP + n] = acc[r];
  }
}

// ---------------------------------------------------------------------------
// Fused attention for one (b, h, 16-row q tile).  128 threads = 4 waves.
// ---------------------------------------------------------------------------
__global__ __launch_bounds__(128) void attn_kernel(
    const __bf16* __restrict__ qb, const __bf16* __restrict__ kb,
    const __bf16* __restrict__ vt, const float* __restrict__ qe,
    const int* __restrict__ mask, const float* __restrict__ emb_v,
    __bf16* __restrict__ ctx) {
  __shared__ __align__(16) float  sc[16][SEQ];     // 32 KB scores / exp
  __shared__ __align__(16) __bf16 at[16][SEQ];     // 16 KB bf16 attn
  __shared__ float wbins[16][NRELP];               //  3 KB bucket weights
  __shared__ float red[2][16][8];                  //  1 KB reductions

  const int blk  = blockIdx.x;        // (b*H + h)*32 + qt
  const int qt   = blk & 31;
  const int bh   = blk >> 5;
  const int b    = bh >> 3;
  const int h    = bh & 7;
  const int q0   = qt * 16;
  const int tid  = threadIdx.x;
  const int wave = tid >> 5;
  const int l    = tid & 31;

  for (int i = tid; i < 16 * NRELP; i += 128) (&wbins[0][0])[i] = 0.0f;

  const __bf16* qbase = qb + ((size_t)bh * SEQ + q0) * DKH;
  const v16bf aq0 = frag_a(qbase, DKH);
  const v16bf aq1 = frag_a(qbase + 32, DKH);
  const float scale = 0.125f;  // 1/sqrt(64)

  // ---- Phase 1: scores -> LDS (WMMA + qe[mask] bias + masking + scale) ----
  for (int i = 0; i < 8; ++i) {
    const int kt = wave * 8 + i;
    const __bf16* kbase = kb + ((size_t)bh * SEQ + kt * 16) * DKH;
    v8f acc = {};
    acc = wmma_bf16(aq0, frag_b(kbase, DKH), acc);
    acc = wmma_bf16(aq1, frag_b(kbase + 32, DKH), acc);
    const int n    = l & 15;
    const int kcol = kt * 16 + n;
#pragma unroll
    for (int r = 0; r < 8; ++r) {
      const int m    = r + ((l >> 4) << 3);
      const int qrow = q0 + m;
      const int rel  = mask[((size_t)b * SEQ + qrow) * SEQ + kcol];
      float s = NEGV;
      if (rel > 0)
        s = (acc[r] + qe[((size_t)bh * SEQ + qrow) * NRELP + rel]) * scale;
      sc[m][kcol] = s;
    }
  }
  __syncthreads();

  // ---- Phase 2: softmax + relative-bucket accumulation ----
  const int row = tid & 15;
  const int seg = tid >> 4;       // 8 segments x 64 columns
  const int c0  = seg * 64;
  float mx = -3.0e38f;
  for (int c = 0; c < 64; ++c) mx = fmaxf(mx, sc[row][c0 + c]);
  red[0][row][seg] = mx;
  __syncthreads();
  float rmx = red[0][row][0];
  for (int s2 = 1; s2 < 8; ++s2) rmx = fmaxf(rmx, red[0][row][s2]);
  float lsum = 0.0f;
  for (int c = 0; c < 64; ++c) {
    const float e = __expf(sc[row][c0 + c] - rmx);
    sc[row][c0 + c] = e;
    lsum += e;
  }
  red[1][row][seg] = lsum;
  __syncthreads();
  float rs = 0.0f;
  for (int s2 = 0; s2 < 8; ++s2) rs += red[1][row][s2];
  const float inv = 1.0f / rs;
  for (int c = 0; c < 64; ++c) {
    const float a = sc[row][c0 + c] * inv;
    at[row][c0 + c] = (__bf16)a;
    const int rel = mask[((size_t)b * SEQ + q0 + row) * SEQ + c0 + c];
    if (rel > 0) atomicAdd(&wbins[row][rel], a);
  }
  __syncthreads();

  // ---- Phase 3: attn @ v (WMMA, A from LDS) + bucket @ emb_v ----
  v8f acc = {};
  const __bf16* vtb = vt + ((size_t)bh * DKH + wave * 16) * SEQ;
  for (int kk = 0; kk < SEQ; kk += 32) {
    v16bf af = frag_a(&at[0][kk], SEQ);
    v16bf bf = frag_b(vtb + kk, SEQ);
    acc = wmma_bf16(af, bf, acc);
  }
  const int n = l & 15;
  const int d = wave * 16 + n;
#pragma unroll
  for (int r = 0; r < 8; ++r) {
    const int m = r + ((l >> 4) << 3);
    float v = acc[r];
    for (int rr = 1; rr < NREL; ++rr)
      v += wbins[m][rr] * emb_v[rr * DKH + d];
    ctx[(((size_t)b * SEQ + q0 + m) * DMODEL) + h * DKH + d] = (__bf16)v;
  }
}

// ---------------------------------------------------------------------------
// Host-side orchestration
// ---------------------------------------------------------------------------
extern "C" void kernel_launch(void* const* d_in, const int* in_sizes, int n_in,
                              void* d_out, int out_size, void* d_ws, size_t ws_size,
                              hipStream_t stream) {
  const float* query = (const float*)d_in[0];
  const float* key   = (const float*)d_in[1];
  const float* value = (const float*)d_in[2];
  const int*   mask  = (const int*)  d_in[3];
  const float* Wq    = (const float*)d_in[4];
  const float* bq    = (const float*)d_in[5];
  const float* Wk    = (const float*)d_in[6];
  const float* bk    = (const float*)d_in[7];
  const float* Wv    = (const float*)d_in[8];
  const float* bv    = (const float*)d_in[9];
  const float* Wo    = (const float*)d_in[10];
  const float* bo    = (const float*)d_in[11];
  const float* emb_k = (const float*)d_in[12];
  const float* emb_v = (const float*)d_in[13];
  float* out = (float*)d_out;

  const size_t M  = (size_t)BATCH * SEQ;       // 2048
  const size_t NX = M * DMODEL;                // 1,048,576
  const size_t NW = (size_t)DMODEL * DMODEL;   // 262,144

  char* ws = (char*)d_ws;
  auto carve = [&](size_t bytes) {
    char* p = ws;
    ws += (bytes + 255) & ~(size_t)255;
    return p;
  };
  __bf16* xqb   = (__bf16*)carve(NX * 2);
  __bf16* xkb   = (__bf16*)carve(NX * 2);
  __bf16* xvb   = (__bf16*)carve(NX * 2);
  __bf16* Wqt   = (__bf16*)carve(NW * 2);
  __bf16* Wkt   = (__bf16*)carve(NW * 2);
  __bf16* Wvt   = (__bf16*)carve(NW * 2);
  __bf16* Wot   = (__bf16*)carve(NW * 2);
  __bf16* qb    = (__bf16*)carve(NX * 2);      // [b,h,s,d]
  __bf16* kb    = (__bf16*)carve(NX * 2);      // [b,h,s,d]
  __bf16* vt    = (__bf16*)carve(NX * 2);      // [b,h,d,s]
  __bf16* embkb = (__bf16*)carve((size_t)NRELP * DKH * 2);
  float*  qebuf = (float*) carve((size_t)BATCH * NHEAD * SEQ * NRELP * 4);
  __bf16* ctx   = (__bf16*)carve(NX * 2);

  // 1) fp32 -> bf16 conversions
  {
    dim3 blk(256), grd((NX + 255) / 256);
    cvt_bf16_kernel<<<grd, blk, 0, stream>>>(query, xqb, (int)NX);
    cvt_bf16_kernel<<<grd, blk, 0, stream>>>(key,   xkb, (int)NX);
    cvt_bf16_kernel<<<grd, blk, 0, stream>>>(value, xvb, (int)NX);
  }
  // 2) weight transposes (N-major for the WMMA B operand)
  {
    dim3 blk(256), grd(DMODEL);
    transpose_w_kernel<<<grd, blk, 0, stream>>>(Wq, Wqt);
    transpose_w_kernel<<<grd, blk, 0, stream>>>(Wk, Wkt);
    transpose_w_kernel<<<grd, blk, 0, stream>>>(Wv, Wvt);
    transpose_w_kernel<<<grd, blk, 0, stream>>>(Wo, Wot);
  }
  // 3) padded bf16 emb_k
  emb_pad_kernel<<<dim3((NRELP * DKH + 255) / 256), dim3(256), 0, stream>>>(emb_k, embkb);

  // 4) Q/K/V projections (TDM-staged WMMA GEMM, 128x128 block tiles)
  {
    dim3 blk(256), grd(DMODEL / 128, M / 128);
    gemm_tile_kernel<0><<<grd, blk, 0, stream>>>(xqb, Wqt, bq, qb, (int)M, DMODEL, DMODEL);
    gemm_tile_kernel<0><<<grd, blk, 0, stream>>>(xkb, Wkt, bk, kb, (int)M, DMODEL, DMODEL);
    gemm_tile_kernel<1><<<grd, blk, 0, stream>>>(xvb, Wvt, bv, vt, (int)M, DMODEL, DMODEL);
  }
  // 5) qe = q @ emb_k^T (WMMA, N padded to 48)
  {
    dim3 blk(32), grd(NRELP / 16, (BATCH * NHEAD * SEQ) / 16);
    qe_kernel<<<grd, blk, 0, stream>>>(qb, embkb, qebuf);
  }
  // 6) fused attention (WMMA scores, LDS softmax, WMMA attn@v)
  attn_kernel<<<dim3(BATCH * NHEAD * (SEQ / 16)), dim3(128), 0, stream>>>(
      qb, kb, vt, qebuf, mask, emb_v, ctx);
  // 7) output projection -> f32 d_out
  {
    dim3 blk(256), grd(DMODEL / 128, M / 128);
    gemm_tile_kernel<2><<<grd, blk, 0, stream>>>(ctx, Wot, bo, out, (int)M, DMODEL, DMODEL);
  }
}
